// Net_7215545057450
// MI455X (gfx1250) — matrix operand order repaired and checked by reading, not compiled, not run on hardware
//
#include <hip/hip_runtime.h>
#include <hip/hip_bf16.h>

#define NN 131072
#define EE 2097152
#define GG 512

typedef __attribute__((ext_vector_type(16))) _Float16 v16h;
typedef __attribute__((ext_vector_type(8)))  float    v8f;
typedef __attribute__((ext_vector_type(8)))  unsigned v8u;

// ---------------- utility kernels ----------------

__global__ __launch_bounds__(256) void k_zero_f32(float* p, long n) {
  long i = (long)blockIdx.x * blockDim.x + threadIdx.x;
  if (i < n) p[i] = 0.0f;
}

// degree accumulation on dst (edge weight 1), then dis = rsqrt(deg+1)
__global__ __launch_bounds__(256) void k_deg(const int* __restrict__ dst, float* __restrict__ deg) {
  int e = blockIdx.x * blockDim.x + threadIdx.x;
  if (e < EE) atomicAdd(&deg[dst[e]], 1.0f);
}

__global__ __launch_bounds__(256) void k_dis(float* deg) {
  int n = blockIdx.x * blockDim.x + threadIdx.x;
  if (n < NN) deg[n] = rsqrtf(deg[n] + 1.0f);
}

// first-layer input (N x 2 f32) -> padded f16 A buffer (N x 32)
__global__ __launch_bounds__(256) void k_convert_input(const float* __restrict__ x, _Float16* __restrict__ xh) {
  long i = (long)blockIdx.x * blockDim.x + threadIdx.x;
  if (i >= (long)NN * 32) return;
  int  k = (int)(i & 31);
  long n = i >> 5;
  xh[i] = (k < 2) ? (_Float16)x[n * 2 + k] : (_Float16)0.0f;
}

// Pack W[fin][fout] (f32) into B-fragment order so each lane's 8 B dwords are
// contiguous: wfrag[((nt*KB + kb)*2 + half)*128 + n16*8 + i] = {W[k][n], W[k+1][n]}
// with k = kb*32 + half*16 + 2*i, n = nt*16 + n16 (zero padded past fin).
__global__ __launch_bounds__(256) void k_pack_w(const float* __restrict__ W, unsigned* __restrict__ wfrag,
                                                int fin, int fout, int Kpad) {
  int KB = Kpad >> 5;
  int total = (Kpad >> 1) * fout;          // = (fout/16)*KB*2*16*8
  int j = blockIdx.x * blockDim.x + threadIdx.x;
  if (j >= total) return;
  int i    = j & 7;
  int n16  = (j >> 3) & 15;
  int hf   = (j >> 7) & 1;
  int rest = j >> 8;
  int kb   = rest % KB;
  int nt   = rest / KB;
  int k0 = kb * 32 + hf * 16 + 2 * i;
  int n  = nt * 16 + n16;
  union { _Float16 h[2]; unsigned u; } p;
  p.h[0] = (k0     < fin) ? (_Float16)W[(k0    ) * fout + n] : (_Float16)0.0f;
  p.h[1] = (k0 + 1 < fin) ? (_Float16)W[(k0 + 1) * fout + n] : (_Float16)0.0f;
  wfrag[j] = p.u;
}

// ---------------- WMMA GEMM: h = xh @ W  (f16 in, f32 acc) ----------------
// Templated on shape so K/N loops fully unroll and B loads are contiguous b128s.
// One wave per 16-row M tile; A fragments for all K blocks held in registers
// and reused across all FOUT/16 output tiles.
template <int KPAD, int FOUT>
__global__ __launch_bounds__(256) void k_gemm_wmma(const _Float16* __restrict__ xh,
                                                   const unsigned* __restrict__ wfrag,
                                                   float* __restrict__ h) {
  constexpr int KB = KPAD / 32;
  constexpr int NT = FOUT / 16;
  int lane = threadIdx.x & 31;
  int wv   = threadIdx.x >> 5;
  int mtile = blockIdx.x * 8 + wv;          // 8 waves/block, 16 rows/wave
  int half  = lane >> 4;                    // 0: lanes 0-15, 1: lanes 16-31
  int col0  = lane & 15;
  const _Float16* arow = xh + ((long)mtile * 16 + col0) * KPAD;

  // A fragment, contiguous k order matching packed B: VGPR i holds K = kb*32 + 16*half + 2*i (+1)
  v8u au[KB];
#pragma unroll
  for (int kb = 0; kb < KB; ++kb) {
#pragma unroll
    for (int i = 0; i < 8; ++i) {
      int ka = kb * 32 + (half << 4) + (i << 1);
      au[kb][i] = *(const unsigned*)(arow + ka);
    }
  }

#pragma unroll
  for (int nt = 0; nt < NT; ++nt) {
    v8f c = {};
#pragma unroll
    for (int kb = 0; kb < KB; ++kb) {
      // B fragment: 8 contiguous dwords, 32B aligned per lane
      const unsigned* bp = wfrag + ((((nt * KB + kb) * 2 + half) * 16 + col0) << 3);
      v8u bu = *(const v8u*)bp;
      v16h a = __builtin_bit_cast(v16h, au[kb]);
      v16h b = __builtin_bit_cast(v16h, bu);
      c = __builtin_amdgcn_wmma_f32_16x16x32_f16(false, a, false, b, (short)0, c, false, false);
    }
#pragma unroll
    for (int r = 0; r < 8; ++r) {
      long row = (long)mtile * 16 + (half << 3) + r;   // C/D: VGPR r -> M = r + 8*half
      h[row * FOUT + nt * 16 + col0] = c[r];
    }
  }
}

// agg = h * dis^2 + bias   (self-loop term + bias, initializes accumulator)
__global__ __launch_bounds__(256) void k_init_agg(const float* __restrict__ h,
                                                  const float* __restrict__ dis,
                                                  const float* __restrict__ bias,
                                                  float* __restrict__ agg, int fout) {
  long i = (long)blockIdx.x * blockDim.x + threadIdx.x;
  if (i >= (long)NN * fout) return;
  long n = i / fout;
  int  c = (int)(i - n * fout);
  float d = dis[n];
  agg[i] = h[i] * d * d + bias[c];
}

// agg[dst] += h[src] * dis[src]*dis[dst] ; one thread per (edge, 4 channels)
__global__ __launch_bounds__(256) void k_scatter(const int* __restrict__ src,
                                                 const int* __restrict__ dst,
                                                 const float* __restrict__ dis,
                                                 const float* __restrict__ h,
                                                 float* __restrict__ agg, int fout) {
  long t = (long)blockIdx.x * blockDim.x + threadIdx.x;
  int groups = fout >> 2;
  long e = t / groups;
  if (e >= EE) return;
  // edge index arrays stream sequentially: prefetch ahead (global_prefetch_b8)
  __builtin_prefetch(&src[e + 4096], 0, 1);
  __builtin_prefetch(&dst[e + 4096], 0, 1);
  int c = (int)(t - e * groups) << 2;
  int s = src[e], d = dst[e];
  float nrm = dis[s] * dis[d];
  const float4 hv = *(const float4*)(h + (long)s * fout + c);
  float* ap = agg + (long)d * fout + c;
  atomicAdd(ap + 0, hv.x * nrm);
  atomicAdd(ap + 1, hv.y * nrm);
  atomicAdd(ap + 2, hv.z * nrm);
  atomicAdd(ap + 3, hv.w * nrm);
}

// ReLU in place; optionally also emit next-layer padded f16 A buffer
__global__ __launch_bounds__(256) void k_relu_convert(float* __restrict__ agg,
                                                      _Float16* __restrict__ xh,
                                                      int fout, int Kpadn) {
  long i = (long)blockIdx.x * blockDim.x + threadIdx.x;
  if (Kpadn > 0) {
    if (i >= (long)NN * Kpadn) return;
    long n = i / Kpadn;
    int  k = (int)(i - n * Kpadn);
    if (k < fout) {
      long idx = n * fout + k;
      float v = fmaxf(agg[idx], 0.0f);
      agg[idx] = v;
      xh[i] = (_Float16)v;
    } else {
      xh[i] = (_Float16)0.0f;
    }
  } else {
    if (i >= (long)NN * fout) return;
    agg[i] = fmaxf(agg[i], 0.0f);
  }
}

// segment_max pooling: values are post-ReLU (>=0), so int-compare atomicMax is monotone
__global__ __launch_bounds__(256) void k_pool(const float* __restrict__ x,
                                              const int* __restrict__ batch,
                                              float* __restrict__ g) {
  long i = (long)blockIdx.x * blockDim.x + threadIdx.x;
  if (i >= (long)NN * 128) return;
  long n = i >> 7;
  int  c = (int)(i & 127);
  int  b = batch[n];
  atomicMax((int*)&g[b * 128 + c], __float_as_int(x[i]));
}

__global__ __launch_bounds__(256) void k_mlp1(const float* __restrict__ g, const float* __restrict__ W,
                                              const float* __restrict__ b, float* __restrict__ h1) {
  int i = blockIdx.x * blockDim.x + threadIdx.x;
  if (i >= GG * 64) return;
  int gi = i >> 6, j = i & 63;
  float acc = b[j];
  for (int k = 0; k < 128; ++k) acc += g[gi * 128 + k] * W[k * 64 + j];
  h1[i] = fmaxf(acc, 0.0f);
}

__global__ __launch_bounds__(256) void k_mlp2(const float* __restrict__ h1, const float* __restrict__ W,
                                              const float* __restrict__ b, float* __restrict__ out) {
  int i = blockIdx.x * blockDim.x + threadIdx.x;
  if (i >= GG * 10) return;
  int gi = i / 10, j = i - gi * 10;
  float acc = b[j];
  for (int k = 0; k < 64; ++k) acc += h1[gi * 64 + k] * W[k * 10 + j];
  out[i] = acc;
}

// ---------------- driver ----------------

extern "C" void kernel_launch(void* const* d_in, const int* in_sizes, int n_in,
                              void* d_out, int out_size, void* d_ws, size_t ws_size,
                              hipStream_t stream) {
  const float* x0    = (const float*)d_in[0];
  const int*   edge  = (const int*)d_in[1];
  const int*   src   = edge;
  const int*   dst   = edge + EE;
  const int*   batch = (const int*)d_in[2];
  const float* Wl1   = (const float*)d_in[15];
  const float* bl1   = (const float*)d_in[16];
  const float* Wl2   = (const float*)d_in[17];
  const float* bl2   = (const float*)d_in[18];
  float* out = (float*)d_out;

  // workspace layout (256B aligned slabs)
  char* ws = (char*)d_ws;
  float*    dis   = (float*)(ws);                                   //  0.5 MB
  float*    bufH  = (float*)(ws + 524288ull);                       // 64 MB  (h = xW)
  float*    bufA  = (float*)(ws + 524288ull + 67108864ull);         // 64 MB  (agg / activations)
  _Float16* xh    = (_Float16*)(ws + 524288ull + 2ull*67108864ull); // 24 MB  (padded f16 A)
  unsigned* wfrag = (unsigned*)(ws + 524288ull + 2ull*67108864ull + 25165824ull);
  float*    g     = (float*)((char*)wfrag + 32768ull);
  float*    h1    = (float*)((char*)g + 262144ull);

  const int DIMS[7] = {2, 16, 32, 48, 64, 96, 128};
  const int KPAD[6] = {32, 32, 32, 64, 64, 96};

  // degrees -> dis = rsqrt(deg+1)
  k_zero_f32<<<(NN + 255) / 256, 256, 0, stream>>>(dis, (long)NN);
  k_deg<<<(EE + 255) / 256, 256, 0, stream>>>(dst, dis);
  k_dis<<<(NN + 255) / 256, 256, 0, stream>>>(dis);

  // input -> padded f16
  {
    long tot = (long)NN * 32;
    k_convert_input<<<(unsigned)((tot + 255) / 256), 256, 0, stream>>>(x0, xh);
  }

  for (int l = 0; l < 6; ++l) {
    int fin = DIMS[l], fout = DIMS[l + 1], Kpad = KPAD[l];
    const float* W = (const float*)d_in[3 + 2 * l];
    const float* b = (const float*)d_in[4 + 2 * l];

    int pt = (Kpad >> 1) * fout;
    k_pack_w<<<(pt + 255) / 256, 256, 0, stream>>>(W, wfrag, fin, fout, Kpad);

    switch (l) {
      case 0: k_gemm_wmma<32,  16><<<NN / 128, 256, 0, stream>>>(xh, wfrag, bufH); break;
      case 1: k_gemm_wmma<32,  32><<<NN / 128, 256, 0, stream>>>(xh, wfrag, bufH); break;
      case 2: k_gemm_wmma<32,  48><<<NN / 128, 256, 0, stream>>>(xh, wfrag, bufH); break;
      case 3: k_gemm_wmma<64,  64><<<NN / 128, 256, 0, stream>>>(xh, wfrag, bufH); break;
      case 4: k_gemm_wmma<64,  96><<<NN / 128, 256, 0, stream>>>(xh, wfrag, bufH); break;
      case 5: k_gemm_wmma<96, 128><<<NN / 128, 256, 0, stream>>>(xh, wfrag, bufH); break;
    }

    long tot = (long)NN * fout;
    k_init_agg<<<(unsigned)((tot + 255) / 256), 256, 0, stream>>>(bufH, dis, b, bufA, fout);

    long st = (long)EE * (fout >> 2);
    k_scatter<<<(unsigned)((st + 255) / 256), 256, 0, stream>>>(src, dst, dis, bufH, bufA, fout);

    int Kpadn = (l < 5) ? KPAD[l + 1] : 0;
    long rt = (long)NN * (Kpadn ? Kpadn : fout);
    k_relu_convert<<<(unsigned)((rt + 255) / 256), 256, 0, stream>>>(bufA, xh, fout, Kpadn);
  }

  // pooling + MLP head
  k_zero_f32<<<(GG * 128 + 255) / 256, 256, 0, stream>>>(g, (long)GG * 128);
  {
    long pt = (long)NN * 128;
    k_pool<<<(unsigned)((pt + 255) / 256), 256, 0, stream>>>(bufA, batch, g);
  }
  k_mlp1<<<(GG * 64 + 255) / 256, 256, 0, stream>>>(g, Wl1, bl1, h1);
  k_mlp2<<<(GG * 10 + 255) / 256, 256, 0, stream>>>(h1, Wl2, bl2, out);
}
